// MultiHeadAttention_57758720196969
// MI455X (gfx1250) — compile-verified, matrix operand order checked by reading, not run on hardware
//
#include <hip/hip_runtime.h>

typedef __attribute__((ext_vector_type(16))) _Float16 v16h;
typedef __attribute__((ext_vector_type(8)))  _Float16 h8;
typedef __attribute__((ext_vector_type(4)))  _Float16 h4;
typedef __attribute__((ext_vector_type(8)))  float    v8f;
typedef __attribute__((ext_vector_type(4)))  float    f4;

#define WMMA_F16(a, b, c) \
  __builtin_amdgcn_wmma_f32_16x16x32_f16(false, (a), false, (b), (short)0, (c), false, false)

namespace {
constexpr int Bn = 8, Sn = 2048, En = 512, Hn = 8, Dn = 64;
}

__device__ __forceinline__ h8 ldh8(const _Float16* p) {
  return *(const h8*)p;                       // b128 load (global or ds)
}
__device__ __forceinline__ v16h cat16(h8 lo, h8 hi) {
  return __builtin_shufflevector(lo, hi, 0, 1, 2, 3, 4, 5, 6, 7,
                                 8, 9, 10, 11, 12, 13, 14, 15);
}

// gfx1250 async global->LDS DMA (ASYNCcnt-tracked), ISA §15.18 opcode 98.
// VDST = per-lane LDS byte address, VADDR = per-lane 64-bit global address.
__device__ __forceinline__ void async_b128(unsigned lds_off, const _Float16* g) {
  asm volatile("global_load_async_to_lds_b128 %0, %1, off"
               :: "v"(lds_off), "v"(g) : "memory");
}
__device__ __forceinline__ void wait_async0() {
  asm volatile("s_wait_asynccnt 0x0" ::: "memory");
}
__device__ __forceinline__ unsigned lds_addr(const void* p) {
  return (unsigned)(unsigned long long)p;     // low 32 bits = LDS byte offset
}

// ---------------------------------------------------------------------------
// Prep kernel A: f32 -> f16 elementwise (x).  4 elems/thread, vectorized.
// ---------------------------------------------------------------------------
__global__ __launch_bounds__(256)
void cvt_f16(const float* __restrict__ in, _Float16* __restrict__ out, int n)
{
  const int i = (blockIdx.x * 256 + threadIdx.x) * 4;
  if (i < n) {
    const f4 v = *(const f4*)(in + i);
    h4 o; o[0] = (_Float16)v.x; o[1] = (_Float16)v.y;
          o[2] = (_Float16)v.z; o[3] = (_Float16)v.w;
    *(h4*)(out + i) = o;
  }
}

// ---------------------------------------------------------------------------
// Prep kernel B: batched f32 [rows,cols] -> f16 transpose [cols,rows].
// ---------------------------------------------------------------------------
__global__ __launch_bounds__(256)
void cvt_t(const float* __restrict__ in, _Float16* __restrict__ out,
           int rows, int cols)
{
  const size_t bo = (size_t)blockIdx.y * rows * cols;
  const int idx = blockIdx.x * 256 + threadIdx.x;
  if (idx < rows * cols) {
    const int r = idx / cols, c = idx % cols;
    out[bo + (size_t)c * rows + r] = (_Float16)in[bo + idx];
  }
}

// ---------------------------------------------------------------------------
// Kernel 1: fused QKV projection (all-f16, pre-transposed weights).
// grid = (S/16, B*H), block = 128 (4 waves); wave = 16-wide D tile.
// Stores q,k row-major [S,D] and V TRANSPOSED [D,S].
// ---------------------------------------------------------------------------
__global__ __launch_bounds__(128)
void mha_qkv(const _Float16* __restrict__ xh,
             const _Float16* __restrict__ WqT,   // [H, D, E]
             const _Float16* __restrict__ WkT,
             const _Float16* __restrict__ WvT,
             _Float16* __restrict__ qo,          // [B,H,S,D], pre-scaled
             _Float16* __restrict__ ko,          // [B,H,S,D]
             _Float16* __restrict__ vTo)         // [B,H,D,S]
{
  const int lane = threadIdx.x & 31;
  const int dt   = threadIdx.x >> 5;
  const int s0   = blockIdx.x * 16;
  const int b    = blockIdx.y / Hn;
  const int h    = blockIdx.y % Hn;
  const int col  = lane & 15;
  const int khi  = (lane & 16) ? 8 : 0;

  const _Float16* xrow = xh + ((size_t)b * Sn + (s0 + col)) * En;
  const _Float16* wqr  = WqT + ((size_t)h * Dn + dt * 16 + col) * En;
  const _Float16* wkr  = WkT + ((size_t)h * Dn + dt * 16 + col) * En;
  const _Float16* wvr  = WvT + ((size_t)h * Dn + dt * 16 + col) * En;

  v8f cq = {}; v8f ck = {}; v8f cv = {};

  for (int e0 = 0; e0 < En; e0 += 32) {
    const v16h a  = cat16(ldh8(xrow + e0 + khi), ldh8(xrow + e0 + 16 + khi));
    const v16h bq = cat16(ldh8(wqr  + e0 + khi), ldh8(wqr  + e0 + 16 + khi));
    const v16h bk = cat16(ldh8(wkr  + e0 + khi), ldh8(wkr  + e0 + 16 + khi));
    const v16h bv = cat16(ldh8(wvr  + e0 + khi), ldh8(wvr  + e0 + 16 + khi));
    cq = WMMA_F16(a, bq, cq);
    ck = WMMA_F16(a, bk, ck);
    cv = WMMA_F16(a, bv, cv);
  }

  const size_t hb  = ((size_t)b * Hn + h) * Sn;
  const size_t vtb = ((size_t)b * Hn + h) * Dn;
#pragma unroll
  for (int r = 0; r < 8; ++r) {
    const size_t idx = (hb + s0 + r + khi) * Dn + dt * 16 + col;
    qo[idx] = (_Float16)(cq[r] * 0.125f);               // fold 1/sqrt(64)
    ko[idx] = (_Float16)ck[r];
    vTo[(vtb + dt * 16 + col) * Sn + s0 + r + khi] = (_Float16)cv[r];
  }
}

// ---------------------------------------------------------------------------
// Kernel 2: causal flash attention.
// grid = (S/64, B*H), block = 128 (4 waves).  Wave w owns 16-query tile
// s0 = 64*blockIdx.x + 16*w.  Each 32-key chunk of K and V^T is staged into
// LDS once per block via GLOBAL_LOAD_ASYNC_TO_LDS_B128 (ASYNCcnt) and shared
// by all 4 waves; fragments are then ds_load_b128 reads.
// Scores computed transposed (K x Q^T) so P^T C-frags re-pack into the P
// A-frag with zero shuffles.
// ---------------------------------------------------------------------------
__global__ __launch_bounds__(128)
void mha_attn(const _Float16* __restrict__ q,
              const _Float16* __restrict__ k,
              const _Float16* __restrict__ vT,   // [B,H,D,S]
              _Float16* __restrict__ att)        // [B,S,H*D] f16
{
  __shared__ _Float16 lk[32 * 64];   // K chunk:   row = key  (32), col = d (64)
  __shared__ _Float16 lv[64 * 32];   // V^T chunk: row = d    (64), col = key (32)

  const int tid  = threadIdx.x;
  const int lane = tid & 31;
  const int w    = tid >> 5;
  const int s0b  = blockIdx.x * 64;
  const int s0   = s0b + w * 16;                 // this wave's query tile
  const int b    = blockIdx.y / Hn;
  const int h    = blockIdx.y % Hn;
  const int col  = lane & 15;
  const int khi  = (lane & 16) ? 8 : 0;

  const _Float16* qb  = q  + ((size_t)b * Hn + h) * Sn * Dn;
  const _Float16* kb  = k  + ((size_t)b * Hn + h) * Sn * Dn;
  const _Float16* vtb = vT + ((size_t)b * Hn + h) * Dn * Sn;

  const unsigned lk0 = lds_addr(&lk[0]);
  const unsigned lv0 = lds_addr(&lv[0]);

  // Q^T B-fragments: lane holds query column s0+col, contiguous in d.
  const _Float16* qrow = qb + (size_t)(s0 + col) * Dn;
  const v16h qf0 = cat16(ldh8(qrow + khi),      ldh8(qrow + 16 + khi));
  const v16h qf1 = cat16(ldh8(qrow + 32 + khi), ldh8(qrow + 48 + khi));

  v8f o0 = {}; v8f o1 = {}; v8f o2 = {}; v8f o3 = {};
  float m_run = -1e30f;
  float l_run = 0.0f;

  const int nch_w   = (s0 / 16 + 2) / 2;         // chunks this wave needs
  const int nch_max = (s0b / 16 + 5) / 2;        // chunks the block stages

  for (int c = 0; c < nch_max; ++c) {
    const int t0 = c * 32;

    // ---- cooperative async stage of K (4KB contiguous) and V^T chunk ----
    async_b128(lk0 + tid * 16,        kb + (size_t)t0 * Dn + tid * 8);
    async_b128(lk0 + tid * 16 + 2048, kb + (size_t)t0 * Dn + (tid + 128) * 8);
    {
      const int r0 = tid >> 2;                   // V^T row (d), 4 lanes/row
      const int c0 = (tid & 3) * 8;              // key offset within row
      async_b128(lv0 + (unsigned)(r0 * 32 + c0) * 2,
                 vtb + (size_t)r0 * Sn + t0 + c0);
      async_b128(lv0 + (unsigned)((r0 + 32) * 32 + c0) * 2,
                 vtb + (size_t)(r0 + 32) * Sn + t0 + c0);
    }
    if (c + 1 < nch_max)                         // overlap HBM->L2 for next chunk
      __builtin_prefetch(kb + (size_t)(t0 + 32) * Dn + tid * 8, 0, 1);

    wait_async0();                               // my DMA done (ASYNCcnt==0)
    __syncthreads();                             // everyone's DMA visible

    if (c < nch_w) {                             // wave-uniform: EXEC stays full
      // ---- scores^T: two 16-key subtiles x two D-halves = 4 WMMAs ----
      v8f sc[2];
      sc[0] = v8f{}; sc[1] = v8f{};
#pragma unroll
      for (int u = 0; u < 2; ++u) {
        const _Float16* krow = &lk[(u * 16 + col) * 64];
        const v16h ka0 = cat16(ldh8(krow + khi),      ldh8(krow + 16 + khi));
        const v16h ka1 = cat16(ldh8(krow + 32 + khi), ldh8(krow + 48 + khi));
        sc[u] = WMMA_F16(ka0, qf0, sc[u]);
        sc[u] = WMMA_F16(ka1, qf1, sc[u]);
      }

      // ---- causal mask + chunk max (per query = per lane column) ----
      float cmax = -1e30f;
#pragma unroll
      for (int u = 0; u < 2; ++u)
#pragma unroll
        for (int r = 0; r < 8; ++r) {
          const int t = t0 + u * 16 + r + khi;
          const float val = (t <= s0 + col) ? sc[u][r] : -1e30f;
          sc[u][r] = val;
          cmax = fmaxf(cmax, val);
        }
      cmax = fmaxf(cmax, __shfl_xor(cmax, 16, 32));
      const float m_new = fmaxf(m_run, cmax);
      const float alpha = __expf(m_run - m_new);

      // ---- exponentiate; P^T C-frags pack into P A-frag directly ----
      float psum = 0.0f;
      v16h pa;
#pragma unroll
      for (int u = 0; u < 2; ++u)
#pragma unroll
        for (int r = 0; r < 8; ++r) {
          const float p = __expf(sc[u][r] - m_new);
          psum += p;
          pa[u * 8 + r] = (_Float16)p;
        }
      psum += __shfl_xor(psum, 16, 32);
      l_run = l_run * alpha + psum;
      m_run = m_new;

      // ---- rescale running O (alpha lives at lane = query index) ----
#pragma unroll
      for (int r = 0; r < 8; ++r) {
        const float ar = __shfl(alpha, r + khi, 32);
        o0[r] *= ar; o1[r] *= ar; o2[r] *= ar; o3[r] *= ar;
      }

      // ---- O += P @ V : V^T rows contiguous in keys -> ds_load_b128 ----
#pragma unroll
      for (int dtile = 0; dtile < 4; ++dtile) {
        const _Float16* vrow = &lv[(dtile * 16 + col) * 32];
        const v16h vf = cat16(ldh8(vrow + khi), ldh8(vrow + 16 + khi));
        if      (dtile == 0) o0 = WMMA_F16(pa, vf, o0);
        else if (dtile == 1) o1 = WMMA_F16(pa, vf, o1);
        else if (dtile == 2) o2 = WMMA_F16(pa, vf, o2);
        else                 o3 = WMMA_F16(pa, vf, o3);
      }
    }
    __syncthreads();                             // reads done before next stage
  }

  // ---- normalize and store head-concatenated f16 [B, S, H*D] ----
#pragma unroll
  for (int r = 0; r < 8; ++r) {
    const float linv = 1.0f / __shfl(l_run, r + khi, 32);
    const size_t ro = ((size_t)b * Sn + s0 + r + khi) * En + h * Dn + col;
    att[ro +  0] = (_Float16)(o0[r] * linv);
    att[ro + 16] = (_Float16)(o1[r] * linv);
    att[ro + 32] = (_Float16)(o2[r] * linv);
    att[ro + 48] = (_Float16)(o3[r] * linv);
  }
}

// ---------------------------------------------------------------------------
// Kernel 3: output projection  Y = att @ Wp + bp  (f32 out, Wp pre-transposed).
// grid = (B*S/16, E/16), block = 32.
// ---------------------------------------------------------------------------
__global__ __launch_bounds__(32)
void mha_proj(const _Float16* __restrict__ att,
              const _Float16* __restrict__ WpT,   // [E, E] transposed
              const float* __restrict__ bp,
              float* __restrict__ out)
{
  const int lane = threadIdx.x & 31;
  const int col  = lane & 15;
  const int khi  = (lane & 16) ? 8 : 0;
  const int m0   = blockIdx.x * 16;
  const int n0   = blockIdx.y * 16;

  const _Float16* arow = att + (size_t)(m0 + col) * En;
  const _Float16* wrow = WpT + (size_t)(n0 + col) * En;
  v8f acc = {};

  for (int e0 = 0; e0 < En; e0 += 32) {
    const v16h a  = cat16(ldh8(arow + e0 + khi), ldh8(arow + e0 + 16 + khi));
    const v16h bf = cat16(ldh8(wrow + e0 + khi), ldh8(wrow + e0 + 16 + khi));
    acc = WMMA_F16(a, bf, acc);
  }

  const float bias = bp[n0 + col];
#pragma unroll
  for (int r = 0; r < 8; ++r)
    out[(size_t)(m0 + r + khi) * En + n0 + col] = acc[r] + bias;
}

// ---------------------------------------------------------------------------
extern "C" void kernel_launch(void* const* d_in, const int* in_sizes, int n_in,
                              void* d_out, int out_size, void* d_ws, size_t ws_size,
                              hipStream_t stream)
{
  (void)in_sizes; (void)n_in; (void)out_size; (void)ws_size;

  const float* x  = (const float*)d_in[0];
  const float* Wq = (const float*)d_in[1];
  const float* Wk = (const float*)d_in[2];
  const float* Wv = (const float*)d_in[3];
  const float* Wp = (const float*)d_in[4];
  const float* bp = (const float*)d_in[5];
  float* out = (float*)d_out;

  const size_t nx   = (size_t)Bn * Sn * En;            // 8.39M
  const size_t nqkv = (size_t)Bn * Hn * Sn * Dn;       // 8.39M
  const size_t nw   = (size_t)Hn * En * Dn;            // 262144

  _Float16* xh    = (_Float16*)d_ws;
  _Float16* qws   = xh + nx;
  _Float16* kws   = qws + nqkv;
  _Float16* vTws  = kws + nqkv;
  _Float16* attws = vTws + nqkv;
  _Float16* wqT   = attws + nx;
  _Float16* wkT   = wqT + nw;
  _Float16* wvT   = wkT + nw;
  _Float16* wpT   = wvT + nw;

  // ---- prep: f16 conversion + weight transposes ----
  cvt_f16<<<dim3((int)(nx / 4 / 256)), 256, 0, stream>>>(x, xh, (int)nx);
  {
    const int pb = (En * Dn + 255) / 256;
    cvt_t<<<dim3(pb, Hn), 256, 0, stream>>>(Wq, wqT, En, Dn);
    cvt_t<<<dim3(pb, Hn), 256, 0, stream>>>(Wk, wkT, En, Dn);
    cvt_t<<<dim3(pb, Hn), 256, 0, stream>>>(Wv, wvT, En, Dn);
    cvt_t<<<dim3((En * En + 255) / 256, 1), 256, 0, stream>>>(Wp, wpT, En, En);
  }

  // ---- main pipeline ----
  mha_qkv <<<dim3(Sn / 16, Bn * Hn),      128, 0, stream>>>(xh, wqT, wkT, wvT,
                                                            qws, kws, vTws);
  mha_attn<<<dim3(Sn / 64, Bn * Hn),      128, 0, stream>>>(qws, kws, vTws, attws);
  mha_proj<<<dim3(Bn * Sn / 16, En / 16),  32, 0, stream>>>(attws, wpT, bp, out);
}
